// GCN_18992345383444
// MI455X (gfx1250) — matrix operand order repaired and checked by reading, not compiled, not run on hardware
//
#include <hip/hip_runtime.h>
#include <hip/hip_bf16.h>

// ---------------------------------------------------------------------------
// GCN layer: y = D^-1/2 graph^T D^-1/2 (x W) + bias   (N=8192, F=512, fp32)
// MI455X plan: deg pass (HBM) -> hs = bf16(dinv[j]*xW) -> 8192^2x512 GEMM on
// v_wmma_f32_16x16x32_bf16 with double-buffered LDS, async B-tile copies
// (GLOBAL_LOAD_ASYNC_TO_LDS_B128), software-pipelined A-tile fp32->bf16
// transpose via native bf16 cvt, dinv[i] row scale + bias in epilogue.
// ---------------------------------------------------------------------------

#define NN 8192
#define FF 512

typedef __attribute__((ext_vector_type(16))) __bf16          v16bf;
typedef __attribute__((ext_vector_type(2)))  __bf16          v2bf;
typedef __attribute__((ext_vector_type(8)))  float           v8f;
typedef __attribute__((ext_vector_type(16))) unsigned short  v16u;
typedef __attribute__((ext_vector_type(8)))  unsigned short  v8u;
typedef __attribute__((ext_vector_type(4)))  int             v4i;

#if __has_builtin(__builtin_amdgcn_global_load_async_to_lds_b128) && \
    __has_builtin(__builtin_amdgcn_s_wait_asynccnt)
#define USE_ASYNC_LDS 1
typedef __attribute__((address_space(1))) v4i gv4i;   // global int4
typedef __attribute__((address_space(3))) v4i lv4i;   // LDS int4
#else
#define USE_ASYNC_LDS 0
#endif

// pack two fp32 -> two bf16 in one dword (lo = a, hi = b), native hw cvt
__device__ __forceinline__ unsigned int pk_bf16(float a, float b) {
  v2bf r;
  r.x = (__bf16)a;
  r.y = (__bf16)b;
  return __builtin_bit_cast(unsigned int, r);
}

// 16B global -> LDS copy: async (ASYNCcnt) when available, sync fallback.
__device__ __forceinline__ void cp16_to_lds(const unsigned short* g,
                                            unsigned short* l) {
#if USE_ASYNC_LDS
  __builtin_amdgcn_global_load_async_to_lds_b128((gv4i*)g, (lv4i*)l, 0, 0);
#else
  *(v8u*)l = *(const v8u*)g;
#endif
}

__device__ __forceinline__ void wait_async_lds() {
#if USE_ASYNC_LDS
  __builtin_amdgcn_s_wait_asynccnt(0);
#endif
}

// --------------------------- kernel 1: zero deg ----------------------------
__global__ __launch_bounds__(256) void gcn_zero(float* __restrict__ deg) {
  deg[blockIdx.x * 256 + threadIdx.x] = 0.0f;
}

// ------------------- kernel 2: deg[i] = sum_j graph[j,i] -------------------
__global__ __launch_bounds__(256) void gcn_deg(const float* __restrict__ graph,
                                               float* __restrict__ deg) {
  const int c  = blockIdx.x * 256 + threadIdx.x;
  const int r0 = blockIdx.y * 128;
  float s = 0.0f;
#pragma unroll 8
  for (int r = 0; r < 128; ++r)
    s += graph[(size_t)(r0 + r) * NN + c];
  atomicAdd(&deg[c], s);
}

// --------- kernel 3: hs[j,f] = bf16( dinv[j] * (x @ weight)[j,f] ) ---------
__global__ __launch_bounds__(256) void gcn_hs(const float* __restrict__ x,
                                              const float* __restrict__ w,
                                              const float* __restrict__ deg,
                                              unsigned short* __restrict__ hs) {
  __shared__ float xs[16][FF];
  const int tx = threadIdx.x;
  const int j0 = blockIdx.x * 16;
#pragma unroll
  for (int t = 0; t < 32; ++t) {
    int idx = tx + t * 256;
    int r = idx >> 9, c = idx & (FF - 1);
    xs[r][c] = x[(size_t)(j0 + r) * FF + c];
  }
  __syncthreads();

  const int f0 = tx, f1 = tx + 256;
  float acc0[16] = {}, acc1[16] = {};
  for (int k = 0; k < FF; ++k) {
    float w0 = w[(size_t)k * FF + f0];
    float w1 = w[(size_t)k * FF + f1];
#pragma unroll
    for (int r = 0; r < 16; ++r) {
      float xv = xs[r][k];
      acc0[r] += xv * w0;
      acc1[r] += xv * w1;
    }
  }
#pragma unroll
  for (int r = 0; r < 16; ++r) {
    float dinv = 1.0f / sqrtf(deg[j0 + r]);
    unsigned int p = pk_bf16(acc0[r] * dinv, acc1[r] * dinv);
    hs[(size_t)(j0 + r) * FF + f0] = (unsigned short)p;
    hs[(size_t)(j0 + r) * FF + f1] = (unsigned short)(p >> 16);
  }
}

// ------------- kernel 4: y = dinv[i]*(graph^T @ hs) + bias -----------------
// Block tile 128(M) x 64(N), K-step 32, 8 waves = 4(M) x 2(N),
// wave tile 32x32 = 2x2 wmma frags. Double-buffered LDS pipeline.
#define BM 128
#define BN 64
#define BK 32
#define LDA 40   // ushorts/row: 80B, multiple of 16B
#define LDB 80   // ushorts/row: 160B, multiple of 16B

// async-stage the 32x64 bf16 B tile (one b128 per thread)
__device__ __forceinline__ void stageB(const unsigned short* __restrict__ hs,
                                       unsigned short (*Bs)[LDB],
                                       int k0, int nb0, int tx) {
  int r  = tx >> 3;        // 0..31
  int c8 = (tx & 7) * 8;   // 0,8,...,56
  cp16_to_lds(hs + (size_t)(k0 + r) * FF + nb0 + c8, &Bs[r][c8]);
}

// convert a staged float4 (column group sc4, row r) and scatter-transpose
__device__ __forceinline__ void storeA(unsigned short (*As)[LDA],
                                       const float4& g, int sc4, int r) {
  unsigned int p01 = pk_bf16(g.x, g.y);
  unsigned int p23 = pk_bf16(g.z, g.w);
  As[4 * sc4 + 0][r] = (unsigned short)p01;
  As[4 * sc4 + 1][r] = (unsigned short)(p01 >> 16);
  As[4 * sc4 + 2][r] = (unsigned short)p23;
  As[4 * sc4 + 3][r] = (unsigned short)(p23 >> 16);
}

extern "C" __global__ __launch_bounds__(256)
void gcn_gemm(const float* __restrict__ graph,
              const unsigned short* __restrict__ hs,
              const float* __restrict__ deg,
              const float* __restrict__ bias,
              float* __restrict__ y) {
  __shared__ unsigned short As[2][BM][LDA];  // As[b][i_local][k_local]
  __shared__ unsigned short Bs[2][BK][LDB];  // Bs[b][k_local][n_local]

  const int tx   = threadIdx.x;
  const int lane = tx & 31;
  const int wave = tx >> 5;
  const int wm   = wave & 3;      // M wave coord -> 32 rows
  const int wn   = wave >> 2;     // N wave coord -> 32 cols
  const int half = lane >> 4;
  const int l16  = lane & 15;

  const int ib0 = blockIdx.y * BM;
  const int nb0 = blockIdx.x * BN;

  // A-staging coords: thread handles float4 (r = k_local, sc4 = col group)
  const int sr  = tx >> 5;        // 0..7  (+8*t)
  const int sc4 = tx & 31;        // float4 column group 0..31

  v8f acc[2][2] = {};

  // ---- prologue: stage tile k0=0 into buffer 0 ----
  stageB(hs, Bs[0], 0, nb0, tx);
#pragma unroll
  for (int t = 0; t < 4; ++t) {
    int r = sr + t * 8;
    const float4 g = *(const float4*)(graph + (size_t)r * NN + ib0 + 4 * sc4);
    storeA(As[0], g, sc4, r);
  }
  wait_async_lds();
  __syncthreads();

  int buf = 0;
  for (int k0 = 0; k0 < NN; k0 += BK) {
    const int nk = k0 + BK;
    const bool has_next = (nk < NN);

    // ---- issue next tile's loads first (latency hidden by WMMAs below) ----
    float4 areg[4];
    if (has_next) {
      stageB(hs, Bs[buf ^ 1], nk, nb0, tx);   // async, no VGPR landing
#pragma unroll
      for (int t = 0; t < 4; ++t)
        areg[t] = *(const float4*)(graph + (size_t)(nk + sr + t * 8) * NN +
                                   ib0 + 4 * sc4);
    }
    // speculative prefetch two tiles ahead (clamped to current tile if out of range)
    {
      int pkt = (k0 + 2 * BK < NN) ? (k0 + 2 * BK) : k0;
      if (tx < 128)
        __builtin_prefetch(graph + (size_t)(pkt + (tx >> 2)) * NN + ib0 +
                               (tx & 3) * 32, 0, 1);
    }

    // ---- fragments from current buffer (ISA 7.12.2 layouts, wave32) ----
    v16u a[2], b[2];
#pragma unroll
    for (int mt = 0; mt < 2; ++mt) {
      int row = wm * 32 + mt * 16 + l16;
      int kb  = half * 8;
      v8u lo = *(const v8u*)(&As[buf][row][kb]);
      v8u hi = *(const v8u*)(&As[buf][row][kb + 16]);
      a[mt] = __builtin_shufflevector(lo, hi, 0, 1, 2, 3, 4, 5, 6, 7,
                                      8, 9, 10, 11, 12, 13, 14, 15);
    }
#pragma unroll
    for (int nt = 0; nt < 2; ++nt) {
      int col = wn * 32 + nt * 16 + l16;
#pragma unroll
      for (int s = 0; s < 16; ++s)
        b[nt][s] = Bs[buf][(s & 8) * 2 + half * 8 + (s & 7)][col];
    }
#pragma unroll
    for (int mt = 0; mt < 2; ++mt)
#pragma unroll
      for (int nt = 0; nt < 2; ++nt)
        acc[mt][nt] = __builtin_amdgcn_wmma_f32_16x16x32_bf16(
            false, __builtin_bit_cast(v16bf, a[mt]),
            false, __builtin_bit_cast(v16bf, b[nt]),
            (short)0, acc[mt][nt], false, false);

    // ---- convert+store next A tile after compute (loadcnt wait lands here) -
    if (has_next) {
#pragma unroll
      for (int t = 0; t < 4; ++t)
        storeA(As[buf ^ 1], areg[t], sc4, sr + t * 8);
    }
    wait_async_lds();   // async B writes for buf^1 landed in LDS
    __syncthreads();    // everyone done reading buf / writing buf^1
    buf ^= 1;
  }

  // ---- epilogue: y = dinv[i]*acc + bias. C/D: N=l16, M=v+8*half ----
#pragma unroll
  for (int mt = 0; mt < 2; ++mt) {
    float dinvv[8];
    int rbase = ib0 + wm * 32 + mt * 16 + 8 * half;
#pragma unroll
    for (int v = 0; v < 8; ++v)
      dinvv[v] = 1.0f / sqrtf(deg[rbase + v]);
#pragma unroll
    for (int nt = 0; nt < 2; ++nt) {
      int col = nb0 + wn * 32 + nt * 16 + l16;
      float bv = bias[col];
#pragma unroll
      for (int v = 0; v < 8; ++v)
        __builtin_nontemporal_store(dinvv[v] * acc[mt][nt][v] + bv,
                                    &y[(size_t)(rbase + v) * FF + col]);
    }
  }
}

// ---------------------------------------------------------------------------
extern "C" void kernel_launch(void* const* d_in, const int* in_sizes, int n_in,
                              void* d_out, int out_size, void* d_ws, size_t ws_size,
                              hipStream_t stream) {
  const float* x      = (const float*)d_in[0];  // [8192,512]
  const float* graph  = (const float*)d_in[1];  // [8192,8192]
  const float* weight = (const float*)d_in[2];  // [512,512]
  const float* bias   = (const float*)d_in[3];  // [512]
  float* y = (float*)d_out;                     // [8192,512]

  float*          deg = (float*)d_ws;                            // 32 KB
  unsigned short* hs  = (unsigned short*)((char*)d_ws + 32768);  // 8 MB bf16

  gcn_zero<<<NN / 256, 256, 0, stream>>>(deg);
  gcn_deg <<<dim3(NN / 256, NN / 128), 256, 0, stream>>>(graph, deg);
  gcn_hs  <<<NN / 16, 256, 0, stream>>>(x, weight, deg, hs);
  gcn_gemm<<<dim3(FF / BN, NN / BM), 256, 0, stream>>>(graph, hs, deg, bias, y);
}
